// SelfDecoder_90623809946293
// MI455X (gfx1250) — matrix operand order repaired and checked by reading, not compiled, not run on hardware
//
#include <hip/hip_runtime.h>
#include <hip/hip_bf16.h>
#include <stdint.h>

#define DEV __device__ __forceinline__

typedef __attribute__((ext_vector_type(16))) __bf16 v16bf;
typedef __attribute__((ext_vector_type(8)))  float  v8f;

constexpr int Bc = 2, Sc = 2048, Dc = 1024, Hc = 16, HDc = 64, Fc = 4096, Lc = 3;
constexpr int Mc = Bc * Sc;            // 4096 rows
constexpr float EPSc = 1e-5f;
constexpr int LDS_STRIDE = 40;         // 128x32 tile padded: 40 halfwords/row -> 0 bank conflicts

union FragAB { v16bf v; uint4 q[2]; unsigned short us[16]; };
union Acc    { v8f v; float f[8]; };

DEV unsigned short f2bf(float f) {
  union { float f; unsigned int u; } c; c.f = f;
  unsigned int u = c.u;
  unsigned int r = u + 0x7FFFu + ((u >> 16) & 1u);
  return (unsigned short)(r >> 16);
}
DEV float bf2f(unsigned short h) {
  union { float f; unsigned int u; } c; c.u = ((unsigned int)h) << 16;
  return c.f;
}

// CDNA5 async global->LDS copy, 16B per lane, tracked by ASYNCcnt.
// LDS offset = low 32 bits of the generic shared-pointer (aperture-based addr).
DEV void async_ld_b128(unsigned lds_off, const void* gaddr) {
  asm volatile("global_load_async_to_lds_b128 %0, %1, off"
               :: "v"(lds_off), "v"(gaddr) : "memory");
}
DEV void wait_async4() { asm volatile("s_wait_asynccnt 4" ::: "memory"); }
DEV void wait_async0() { asm volatile("s_wait_asynccnt 0" ::: "memory"); }

// A fragment (16x32 bf16): lane holds row (lane&15); us[0..7] = K(half*8 + 0..7),
// us[8..15] = K(half*8 + 16..23).   [CDNA5 ISA "16-bit A-Matrix 16x32"]
DEV FragAB lds_frag_a(const unsigned short* p0) {           // p0 -> row base in LDS
  FragAB f;
  f.q[0] = *(const uint4*)(p0);
  f.q[1] = *(const uint4*)(p0 + 16);
  return f;
}
// B fragment (32x16 bf16), B[k][n] = W[n][k]: lane holds W-row n=(lane&15);
// us[0..7] = K(half*8 + 0..7), us[8..15] = K(half*8 + 16..23).
DEV FragAB lds_frag_b(const unsigned short* p0) {
  FragAB f;
  f.q[0] = *(const uint4*)(p0);
  f.q[1] = *(const uint4*)(p0 + 16);
  return f;
}

// ---------------------------------------------------------------- cast fp32->bf16
__global__ __launch_bounds__(256)
void cast_f32_bf16(const float* __restrict__ in, unsigned short* __restrict__ out, long long n) {
  long long i = ((long long)blockIdx.x * 256 + threadIdx.x) * 4;
  if (i + 3 < n) {
    float4 v = *(const float4*)(in + i);
    out[i]   = f2bf(v.x); out[i+1] = f2bf(v.y);
    out[i+2] = f2bf(v.z); out[i+3] = f2bf(v.w);
  }
}

// ---------------------------------------------------------------- RMSNorm + cast
__global__ __launch_bounds__(256)
void rmsnorm_cast(const float* __restrict__ x, const float* __restrict__ g,
                  unsigned short* __restrict__ h) {
  __shared__ float red[8];
  __shared__ float sscale;
  int row = blockIdx.x;
  int t = threadIdx.x;
  const float* xr = x + (size_t)row * Dc;
  float4 v = *(const float4*)(xr + t * 4);
  float ss = v.x*v.x + v.y*v.y + v.z*v.z + v.w*v.w;
  #pragma unroll
  for (int m = 16; m >= 1; m >>= 1) ss += __shfl_xor(ss, m, 32);
  if ((t & 31) == 0) red[t >> 5] = ss;
  __syncthreads();
  if (t < 8) {
    float s2 = red[t];
    #pragma unroll
    for (int m = 4; m >= 1; m >>= 1) s2 += __shfl_xor(s2, m, 32);
    if (t == 0) sscale = rsqrtf(s2 / (float)Dc + EPSc);
  }
  __syncthreads();
  float sc = sscale;
  float4 gv = *(const float4*)(g + t * 4);
  unsigned short* hr = h + (size_t)row * Dc + t * 4;
  hr[0] = f2bf(v.x * sc * gv.x);
  hr[1] = f2bf(v.y * sc * gv.y);
  hr[2] = f2bf(v.z * sc * gv.z);
  hr[3] = f2bf(v.w * sc * gv.w);
}

// ---------------------------------------------------------------- SwiGLU (in-place ok)
__global__ __launch_bounds__(256)
void swiglu_kernel(const unsigned short* __restrict__ u, const unsigned short* __restrict__ g,
                   unsigned short* __restrict__ out, long long n) {
  long long i = ((long long)blockIdx.x * 256 + threadIdx.x) * 4;
  if (i >= n) return;
  #pragma unroll
  for (int j = 0; j < 4; j++) {
    float uu = bf2f(u[i + j]);
    float gg = bf2f(g[i + j]);
    float s = uu / (1.0f + __expf(-uu));
    out[i + j] = f2bf(s * gg);
  }
}

// ---------------------------------------------------------------- GEMM: Y = A @ W^T
// A: bf16 [M,K] row-major.  W: bf16 [N,K] row-major.
// Block = 256 thr = 8 waves; block tile 128x128, wave tile 64x32 (4x2 WMMA tiles).
// K stepped by 32; A/B 128x32 tiles double-buffered in LDS via async global->LDS
// (ASYNCcnt), fragments served from padded LDS (conflict-free).
// OUTMODE 0: fp32 out + residual add    (Wo / W2 projections)
// OUTMODE 1: bf16 out [M,N]             (Q,K,U,G projections)
// OUTMODE 2: bf16 out transposed per head -> [b,h,d,s]  (V projection)
template<int OUTMODE>
__global__ __launch_bounds__(256)
void gemm_bf16(const unsigned short* __restrict__ A, const unsigned short* __restrict__ W,
               void* __restrict__ outp, const float* __restrict__ res,
               int N, int K) {
  __shared__ __align__(16) unsigned short sA[2][128 * LDS_STRIDE];
  __shared__ __align__(16) unsigned short sB[2][128 * LDS_STRIDE];

  const int tid  = threadIdx.x;
  const int lane = tid & 31;
  const int wave = tid >> 5;
  const int half = lane >> 4;
  const int ln   = lane & 15;
  const int mBlk = blockIdx.y * 128;
  const int nBlk = blockIdx.x * 128;
  const int mW = (wave >> 2) * 64;     // wave row offset inside block tile
  const int nW = (wave & 3) * 32;      // wave col offset inside block tile

  // stage one 128x32 bf16 tile: 512 x 16B chunks, 2 per thread
  auto stage = [&](const unsigned short* gbase, int rowBase, int k, unsigned short* sdst) {
    #pragma unroll
    for (int t = 0; t < 2; t++) {
      int idx = tid + t * 256;
      int row = idx >> 2, ch = idx & 3;
      const unsigned short* g = gbase + (size_t)(rowBase + row) * K + k + ch * 8;
      unsigned lo = (unsigned)(size_t)(sdst + row * LDS_STRIDE + ch * 8);
      async_ld_b128(lo, g);
    }
  };

  Acc acc[4][2];
  #pragma unroll
  for (int i = 0; i < 4; i++)
    #pragma unroll
    for (int j = 0; j < 2; j++) { v8f z = {0,0,0,0,0,0,0,0}; acc[i][j].v = z; }

  const int KT = K >> 5;
  stage(A, mBlk, 0, sA[0]);
  stage(W, nBlk, 0, sB[0]);

  for (int ki = 0; ki < KT; ki++) {
    const int cur = ki & 1;
    if (ki + 1 < KT) {
      stage(A, mBlk, (ki + 1) << 5, sA[cur ^ 1]);
      stage(W, nBlk, (ki + 1) << 5, sB[cur ^ 1]);
      wait_async4();                  // previous batch (this buffer) landed
    } else {
      wait_async0();
    }
    __syncthreads();                  // all waves' tiles visible

    FragAB af[4], bfr[2];
    #pragma unroll
    for (int i = 0; i < 4; i++)
      af[i] = lds_frag_a(&sA[cur][(mW + i * 16 + ln) * LDS_STRIDE + half * 8]);
    #pragma unroll
    for (int j = 0; j < 2; j++)
      bfr[j] = lds_frag_b(&sB[cur][(nW + j * 16 + ln) * LDS_STRIDE + half * 8]);
    #pragma unroll
    for (int i = 0; i < 4; i++)
      #pragma unroll
      for (int j = 0; j < 2; j++)
        acc[i][j].v = __builtin_amdgcn_wmma_f32_16x16x32_bf16(
            false, af[i].v, false, bfr[j].v, (short)0, acc[i][j].v, false, false);
    __syncthreads();                  // reads done before buffer reuse
  }

  // C/D layout: vgpr r -> m = half*8 + r ; n = lane&15   [ISA 32-bit C/D 16x16]
  #pragma unroll
  for (int i = 0; i < 4; i++)
    #pragma unroll
    for (int j = 0; j < 2; j++)
      #pragma unroll
      for (int r = 0; r < 8; r++) {
        int m = mBlk + mW + i * 16 + half * 8 + r;
        int n = nBlk + nW + j * 16 + ln;
        float vv = acc[i][j].f[r];
        if (OUTMODE == 0) {
          float* out = (float*)outp;
          out[(size_t)m * N + n] = res[(size_t)m * N + n] + vv;
        } else if (OUTMODE == 1) {
          unsigned short* out = (unsigned short*)outp;
          out[(size_t)m * N + n] = f2bf(vv);
        } else {
          unsigned short* out = (unsigned short*)outp;
          int b = m / Sc, s = m % Sc, hh = n / HDc, d = n % HDc;
          out[(((size_t)(b * Hc + hh)) * HDc + d) * Sc + s] = f2bf(vv);
        }
      }
}

// ---------------------------------------------------------------- flash attention
// One wave = one (b, h, 16-row q-tile). Online softmax; 32-key tiles.
__global__ __launch_bounds__(256)
void attn_kernel(const unsigned short* __restrict__ Q,
                 const unsigned short* __restrict__ Kb,
                 const unsigned short* __restrict__ Vt,   // [b,h,d,s] bf16
                 unsigned short* __restrict__ O) {
  __shared__ __align__(16) unsigned short ldsP[8][16 * 32];
  int lane = threadIdx.x & 31;
  int wv   = threadIdx.x >> 5;
  int half = lane >> 4;
  int ln   = lane & 15;
  const int QT = Sc / 16;
  int job = blockIdx.x * 8 + wv;
  if (job >= Bc * Hc * QT) return;           // whole-wave uniform
  int qt = job % QT;
  int h  = (job / QT) % Hc;
  int b  = job / (QT * Hc);
  int qbase = qt * 16;

  FragAB qf[2];                              // Q 16x64 split into two 16x32 A-frags
  #pragma unroll
  for (int s = 0; s < 2; s++) {
    const unsigned short* p = Q + ((size_t)(b * Sc + qbase + ln)) * Dc + h * HDc + s * 32 + half * 8;
    qf[s].q[0] = *(const uint4*)p;
    qf[s].q[1] = *(const uint4*)(p + 16);
  }

  Acc o[4];
  #pragma unroll
  for (int t = 0; t < 4; t++) { v8f z = {0,0,0,0,0,0,0,0}; o[t].v = z; }
  float mstat[8], lstat[8];
  #pragma unroll
  for (int r = 0; r < 8; r++) { mstat[r] = -1e30f; lstat[r] = 0.0f; }
  const float scale = 0.125f;                // 1/sqrt(64)

  for (int kb = 0; kb < qbase + 16; kb += 32) {
    Acc st[2];
    #pragma unroll
    for (int t = 0; t < 2; t++) {
      v8f z = {0,0,0,0,0,0,0,0}; st[t].v = z;
      #pragma unroll
      for (int s = 0; s < 2; s++) {          // K^T as B operand: row n = key
        FragAB kf;
        const unsigned short* p =
            Kb + ((size_t)(b * Sc + kb + t * 16 + ln)) * Dc + h * HDc + s * 32 + half * 8;
        kf.q[0] = *(const uint4*)p;
        kf.q[1] = *(const uint4*)(p + 16);
        st[t].v = __builtin_amdgcn_wmma_f32_16x16x32_bf16(
            false, qf[s].v, false, kf.v, (short)0, st[t].v, false, false);
      }
    }
    #pragma unroll
    for (int r = 0; r < 8; r++) {
      int m = qbase + half * 8 + r;
      #pragma unroll
      for (int t = 0; t < 2; t++) {
        int n = kb + t * 16 + ln;
        float vv = st[t].f[r] * scale;
        if (n > m) vv = -1e30f;              // causal mask
        st[t].f[r] = vv;
      }
      float mx = fmaxf(st[0].f[r], st[1].f[r]);
      #pragma unroll
      for (int msk = 8; msk >= 1; msk >>= 1) mx = fmaxf(mx, __shfl_xor(mx, msk, 32));
      float mnew  = fmaxf(mstat[r], mx);
      float alpha = __expf(mstat[r] - mnew);
      mstat[r] = mnew;
      lstat[r] *= alpha;
      #pragma unroll
      for (int t = 0; t < 4; t++) o[t].f[r] *= alpha;
      float psum = 0.0f;
      #pragma unroll
      for (int t = 0; t < 2; t++) {
        float p = __expf(st[t].f[r] - mnew);
        st[t].f[r] = p;
        psum += p;
      }
      #pragma unroll
      for (int msk = 8; msk >= 1; msk >>= 1) psum += __shfl_xor(psum, msk, 32);
      lstat[r] += psum;
    }
    // C-layout -> A-fragment layout via per-wave LDS round trip
    #pragma unroll
    for (int t = 0; t < 2; t++)
      #pragma unroll
      for (int r = 0; r < 8; r++)
        ldsP[wv][(half * 8 + r) * 32 + t * 16 + ln] = f2bf(st[t].f[r]);
    asm volatile("s_wait_dscnt 0" ::: "memory");   // in-wave LDS ordering (CDNA5)
    FragAB pf;
    {
      const unsigned short* p = &ldsP[wv][ln * 32 + half * 8];
      pf.q[0] = *(const uint4*)p;
      pf.q[1] = *(const uint4*)(p + 16);
    }
    #pragma unroll
    for (int t = 0; t < 4; t++) {            // V as B operand from [b,h,d,s]
      FragAB vf;
      const unsigned short* p =
          Vt + (((size_t)(b * Hc + h)) * HDc + t * 16 + ln) * Sc + kb + half * 8;
      vf.q[0] = *(const uint4*)p;
      vf.q[1] = *(const uint4*)(p + 16);
      o[t].v = __builtin_amdgcn_wmma_f32_16x16x32_bf16(
          false, pf.v, false, vf.v, (short)0, o[t].v, false, false);
    }
  }
  #pragma unroll
  for (int t = 0; t < 4; t++)
    #pragma unroll
    for (int r = 0; r < 8; r++) {
      int m = qbase + half * 8 + r;
      int n = h * HDc + t * 16 + ln;
      O[((size_t)(b * Sc + m)) * Dc + n] = f2bf(o[t].f[r] / lstat[r]);
    }
}

// ---------------------------------------------------------------- host launcher
extern "C" void kernel_launch(void* const* d_in, const int* in_sizes, int n_in,
                              void* d_out, int out_size, void* d_ws, size_t ws_size,
                              hipStream_t stream) {
  const float* x_in = (const float*)d_in[0];
  const float* Wq = (const float*)d_in[1];
  const float* Wk = (const float*)d_in[2];
  const float* Wv = (const float*)d_in[3];
  const float* Wo = (const float*)d_in[4];
  const float* W1 = (const float*)d_in[5];
  const float* Wg = (const float*)d_in[6];
  const float* W2 = (const float*)d_in[7];
  const float* g1 = (const float*)d_in[8];
  const float* g2 = (const float*)d_in[9];

  char* ws = (char*)d_ws;
  size_t off = 0;
  auto alloc = [&](size_t bytes) -> void* {
    void* p = ws + off;
    off += (bytes + 255) & ~(size_t)255;
    return p;
  };
  const size_t DD = (size_t)Dc * Dc, FD = (size_t)Fc * Dc;
  float* x            = (float*)alloc((size_t)Mc * Dc * 4);
  unsigned short* wqB = (unsigned short*)alloc(Lc * DD * 2);
  unsigned short* wkB = (unsigned short*)alloc(Lc * DD * 2);
  unsigned short* wvB = (unsigned short*)alloc(Lc * DD * 2);
  unsigned short* woB = (unsigned short*)alloc(Lc * DD * 2);
  unsigned short* w1B = (unsigned short*)alloc(Lc * FD * 2);
  unsigned short* wgB = (unsigned short*)alloc(Lc * FD * 2);
  unsigned short* w2B = (unsigned short*)alloc(Lc * FD * 2);
  unsigned short* hB  = (unsigned short*)alloc((size_t)Mc * Dc * 2);
  unsigned short* qB  = (unsigned short*)alloc((size_t)Mc * Dc * 2);
  unsigned short* kB  = (unsigned short*)alloc((size_t)Mc * Dc * 2);
  unsigned short* vT  = (unsigned short*)alloc((size_t)Mc * Dc * 2);
  unsigned short* oB  = (unsigned short*)alloc((size_t)Mc * Dc * 2);
  unsigned short* uB  = (unsigned short*)alloc((size_t)Mc * Fc * 2);
  unsigned short* gB  = (unsigned short*)alloc((size_t)Mc * Fc * 2);
  unsigned short* hgB = uB;                            // silu(u)*g in place of u

  hipMemcpyAsync(x, x_in, (size_t)Mc * Dc * sizeof(float),
                 hipMemcpyDeviceToDevice, stream);

  auto castw = [&](const float* src, unsigned short* dst, long long n) {
    cast_f32_bf16<<<(unsigned)((n + 1023) / 1024), 256, 0, stream>>>(src, dst, n);
  };
  castw(Wq, wqB, (long long)Lc * DD);
  castw(Wk, wkB, (long long)Lc * DD);
  castw(Wv, wvB, (long long)Lc * DD);
  castw(Wo, woB, (long long)Lc * DD);
  castw(W1, w1B, (long long)Lc * FD);
  castw(Wg, wgB, (long long)Lc * FD);
  castw(W2, w2B, (long long)Lc * FD);

  dim3 blk(256);
  dim3 gD(Dc / 128, Mc / 128);   // N=1024 GEMMs
  dim3 gF(Fc / 128, Mc / 128);   // N=4096 GEMMs

  for (int i = 0; i < Lc; i++) {
    // ---- attention block ----
    rmsnorm_cast<<<Mc, blk, 0, stream>>>(x, g1 + (size_t)i * Dc, hB);
    gemm_bf16<1><<<gD, blk, 0, stream>>>(hB, wqB + (size_t)i * DD, qB, nullptr, Dc, Dc);
    gemm_bf16<1><<<gD, blk, 0, stream>>>(hB, wkB + (size_t)i * DD, kB, nullptr, Dc, Dc);
    gemm_bf16<2><<<gD, blk, 0, stream>>>(hB, wvB + (size_t)i * DD, vT, nullptr, Dc, Dc);
    attn_kernel<<<(Bc * Hc * (Sc / 16)) / 8, blk, 0, stream>>>(qB, kB, vT, oB);
    gemm_bf16<0><<<gD, blk, 0, stream>>>(oB, woB + (size_t)i * DD, x, x, Dc, Dc);
    // ---- SwiGLU FFN block ----
    rmsnorm_cast<<<Mc, blk, 0, stream>>>(x, g2 + (size_t)i * Dc, hB);
    gemm_bf16<1><<<gF, blk, 0, stream>>>(hB, w1B + (size_t)i * FD, uB, nullptr, Fc, Dc);
    gemm_bf16<1><<<gF, blk, 0, stream>>>(hB, wgB + (size_t)i * FD, gB, nullptr, Fc, Dc);
    long long nFF = (long long)Mc * Fc;
    swiglu_kernel<<<(unsigned)((nFF + 1023) / 1024), blk, 0, stream>>>(uB, gB, hgB, nFF);
    float* dst = (i == Lc - 1) ? (float*)d_out : x;
    gemm_bf16<0><<<gD, blk, 0, stream>>>(hgB, w2B + (size_t)i * FD, dst, x, Dc, Fc);
  }
}
